// CaptioningRNN_63436666962431
// MI455X (gfx1250) — matrix operand order repaired and checked by reading, not compiled, not run on hardware
//
#include <hip/hip_runtime.h>

// Problem constants (from reference setup_inputs)
#define NB   128        // batch
#define TT   64         // timesteps
#define DD   512        // x feature dim
#define HH   1280       // hidden
#define KTOT 3072       // DD + HH + HH  (fused K for [xt | h | attn])
#define NOUT 5120       // 4*H gate outputs
#define KTILES (KTOT / 32)     // 96 k-tiles of 32
#define KSPLIT 4               // split-K factor for the recurrent GEMM
#define KTPC  (KTILES / KSPLIT) // 24 k-tiles per chunk

typedef __attribute__((ext_vector_type(16))) __bf16 v16bf;
typedef __attribute__((ext_vector_type(8)))  float  v8f;

// fp32 -> bf16, round-to-nearest-even (matches WMMA bf16 expectations)
__device__ __forceinline__ unsigned short f2bf(float f) {
  unsigned int u = __float_as_uint(f);
  u += 0x7FFFu + ((u >> 16) & 1u);
  return (unsigned short)(u >> 16);
}

// ---- WMMA fragment-native packed layouts (ISA 7.12.2, 16-bit A/B maps) ----
// A (RHS activations), 16x32 tile: lane = (m&15) + 16*((kin>>3)&1),
// within-lane bf16 index j = (kin&7) + 8*((kin>>4)&1). 512 bf16 per tile.
__device__ __forceinline__ unsigned int rhsPackedIdx(int m, int k) {
  int mb   = m >> 4;
  int kt   = k >> 5;
  int kin  = k & 31;
  int lane = (m & 15) + (((kin >> 3) & 1) << 4);
  int j    = (kin & 7) + (((kin >> 4) & 1) << 3);
  return (unsigned int)(((mb * KTILES + kt) << 9) + (lane << 4) + j);
}

// B (weights), 32x16 tile: lane = (n&15) + 16*((kin>>3)&1),
// j = (kin&7) + 8*((kin>>4)&1).
__device__ __forceinline__ unsigned int wPackedIdx(int k, int n) {
  int nb   = n >> 4;
  int kt   = k >> 5;
  int kin  = k & 31;
  int lane = (n & 15) + (((kin >> 3) & 1) << 4);
  int j    = (kin & 7) + (((kin >> 4) & 1) << 3);
  return (unsigned int)(((nb * KTILES + kt) << 9) + (lane << 4) + j);
}

// -------- one-time: pack [Wx; Wh; Wattn] -> bf16 fragment layout ----------
__global__ void pack_weights(const float* __restrict__ Wx,
                             const float* __restrict__ Wh,
                             const float* __restrict__ Wattn,
                             unsigned short* __restrict__ Wp) {
  int idx = blockIdx.x * blockDim.x + threadIdx.x;
  if (idx >= KTOT * NOUT) return;
  int k = idx / NOUT;
  int n = idx - k * NOUT;           // n fastest -> coalesced source reads
  float v;
  if (k < DD)            v = Wx[k * NOUT + n];
  else if (k < DD + HH)  v = Wh[(k - DD) * NOUT + n];
  else                   v = Wattn[(k - DD - HH) * NOUT + n];
  Wp[wPackedIdx(k, n)] = f2bf(v);
}

// -------- init: h0 = c0 = mean over 16 spatial locs of A ------------------
__global__ void init_state(const float* __restrict__ A,
                           float* __restrict__ hst, float* __restrict__ cst,
                           unsigned short* __restrict__ Rp) {
  int idx = blockIdx.x * blockDim.x + threadIdx.x;
  if (idx >= NB * HH) return;
  int n = idx / HH, h = idx - n * HH;
  const float* af = A + (n * HH + h) * 16;
  float s = 0.f;
#pragma unroll
  for (int l = 0; l < 16; ++l) s += af[l];
  s *= 0.0625f;
  hst[idx] = s;
  cst[idx] = s;
  Rp[rhsPackedIdx(n, DD + h)] = f2bf(s);
}

// -------- per step: scaled-dot attention over 16 locations + xt staging ---
// one workgroup per batch element n (256 threads, 8 waves)
__global__ void attn_step(const float* __restrict__ x,
                          const float* __restrict__ A,
                          const float* __restrict__ hst,
                          unsigned short* __restrict__ Rp, int t) {
  __shared__ float sp[256];
  __shared__ float ex[16];
  __shared__ float wv[16];
  int n   = blockIdx.x;
  int tid = threadIdx.x;
  const float* Af = A + n * HH * 16;
  const float* hv = hst + n * HH;

  // scores[l] = scale * sum_h h[h]*Af[h][l]; 16 threads per location
  int l = tid & 15, g = tid >> 4;
  float p = 0.f;
  for (int h = g; h < HH; h += 16) p += hv[h] * Af[h * 16 + l];
  sp[tid] = p;
  __syncthreads();
  if (tid < 16) {
    float s = 0.f;
#pragma unroll
    for (int gg = 0; gg < 16; ++gg) s += sp[gg * 16 + tid];
    sp[tid] = s * 0.027950849718747373f;   // 1/sqrt(H), H=1280
  }
  __syncthreads();
  if (tid < 16) {
    float m = sp[0];
#pragma unroll
    for (int i = 1; i < 16; ++i) m = fmaxf(m, sp[i]);
    ex[tid] = __expf(sp[tid] - m);
  }
  __syncthreads();
  if (tid < 16) {
    float s = 0.f;
#pragma unroll
    for (int i = 0; i < 16; ++i) s += ex[i];
    wv[tid] = ex[tid] / s;
  }
  __syncthreads();

  // attn[h] = sum_l Af[h][l]*w[l]  -> bf16 into packed RHS at k = DD+HH+h
  for (int h = tid; h < HH; h += 256) {
    const float* a = Af + h * 16;
    float acc = 0.f;
#pragma unroll
    for (int ll = 0; ll < 16; ++ll) acc += a[ll] * wv[ll];
    Rp[rhsPackedIdx(n, DD + HH + h)] = f2bf(acc);
  }
  // stage xt slice -> packed RHS at k = 0..DD-1
  const float* xr = x + (n * TT + t) * DD;
  for (int kx = tid; kx < DD; kx += 256)
    Rp[rhsPackedIdx(n, kx)] = f2bf(xr[kx]);
}

// -------- per step: a = [xt|h|attn] @ [Wx;Wh;Wattn], bf16 WMMA, f32 acc ---
// Split-K x4: grid (80, 4), 256 threads. Each WG: 128 rows x 64 cols over
// 24 k-tiles, partials to slice blockIdx.y. 8 waves as 4(M) x 2(N); each
// wave computes 32x32 via 4 wmma accumulators.
union AFrag { uint4 u[2]; v16bf v; };

__global__ void __launch_bounds__(256) gemm_step(
    const unsigned short* __restrict__ Rp,
    const unsigned short* __restrict__ Wp,
    float* __restrict__ aout) {
  int lane  = threadIdx.x & 31;
  int wave  = threadIdx.x >> 5;
  int waveM = wave & 3;
  int waveN = wave >> 2;
  int nb0   = blockIdx.x * 4 + waveN * 2;   // 16-col blocks
  int mb0   = waveM * 2;                    // 16-row blocks
  int kc    = blockIdx.y;                   // split-K chunk

  unsigned int kbase = (unsigned)kc * KTPC * 512u + (unsigned)lane * 16u;
  const unsigned short* Ap0 = Rp + (unsigned)(mb0      * KTILES) * 512u + kbase;
  const unsigned short* Ap1 = Rp + (unsigned)((mb0 + 1)* KTILES) * 512u + kbase;
  const unsigned short* Bp0 = Wp + (unsigned)(nb0      * KTILES) * 512u + kbase;
  const unsigned short* Bp1 = Wp + (unsigned)((nb0 + 1)* KTILES) * 512u + kbase;

  v8f acc00 = {}; v8f acc01 = {}; v8f acc10 = {}; v8f acc11 = {};

#pragma unroll 2
  for (int kt = 0; kt < KTPC; ++kt) {
    // prefetch upcoming weight tiles (stay hot in L0/L2)
    __builtin_prefetch(Bp0 + (kt + 4) * 512, 0, 1);
    __builtin_prefetch(Bp1 + (kt + 4) * 512, 0, 1);
    AFrag a0 = *(const AFrag*)(Ap0 + kt * 512);
    AFrag a1 = *(const AFrag*)(Ap1 + kt * 512);
    AFrag b0 = *(const AFrag*)(Bp0 + kt * 512);
    AFrag b1 = *(const AFrag*)(Bp1 + kt * 512);
    acc00 = __builtin_amdgcn_wmma_f32_16x16x32_bf16(false, a0.v, false, b0.v,
                                                    (short)0, acc00, false, false);
    acc01 = __builtin_amdgcn_wmma_f32_16x16x32_bf16(false, a0.v, false, b1.v,
                                                    (short)0, acc01, false, false);
    acc10 = __builtin_amdgcn_wmma_f32_16x16x32_bf16(false, a1.v, false, b0.v,
                                                    (short)0, acc10, false, false);
    acc11 = __builtin_amdgcn_wmma_f32_16x16x32_bf16(false, a1.v, false, b1.v,
                                                    (short)0, acc11, false, false);
  }

  // C/D layout: lane n = lane&15; VGPR r holds m = r + 8*(lane>>4)
  int half8 = (lane >> 4) << 3;
  int cA = nb0 * 16 + (lane & 15);
  int cB = cA + 16;
  float* slice = aout + (unsigned)kc * (NB * NOUT);
  float* r0 = slice + (unsigned)((mb0 * 16)       + half8) * NOUT;
  float* r1 = slice + (unsigned)(((mb0 + 1) * 16) + half8) * NOUT;
#pragma unroll
  for (int r = 0; r < 8; ++r) {
    r0[r * NOUT + cA] = acc00[r];
    r0[r * NOUT + cB] = acc01[r];
    r1[r * NOUT + cA] = acc10[r];
    r1[r * NOUT + cB] = acc11[r];
  }
}

// -------- per step: sum split-K slices, gates, cell update, emit h_t ------
__global__ void gates_step(const float* __restrict__ aout,
                           const float* __restrict__ b,
                           float* __restrict__ hst, float* __restrict__ cst,
                           unsigned short* __restrict__ Rp,
                           float* __restrict__ out, int t) {
  int idx = blockIdx.x * blockDim.x + threadIdx.x;
  if (idx >= NB * HH) return;
  int n = idx / HH, h = idx - n * HH;
  float ai = b[h];
  float af = b[HH + h];
  float ao = b[2 * HH + h];
  float ag = b[3 * HH + h];
#pragma unroll
  for (int s = 0; s < KSPLIT; ++s) {
    const float* ar = aout + (unsigned)s * (NB * NOUT) + n * NOUT;
    ai += ar[h];
    af += ar[HH + h];
    ao += ar[2 * HH + h];
    ag += ar[3 * HH + h];
  }
  float ig = 1.f / (1.f + __expf(-ai));
  float fg = 1.f / (1.f + __expf(-af));
  float og = 1.f / (1.f + __expf(-ao));
  float gg = tanhf(ag);
  float c  = fg * cst[idx] + ig * gg;
  float hn = og * tanhf(c);
  cst[idx] = c;
  hst[idx] = hn;
  Rp[rhsPackedIdx(n, DD + h)] = f2bf(hn);        // RHS for step t+1
  out[(n * TT + t) * HH + h] = hn;               // (N, T, H)
}

extern "C" void kernel_launch(void* const* d_in, const int* in_sizes, int n_in,
                              void* d_out, int out_size, void* d_ws, size_t ws_size,
                              hipStream_t stream) {
  (void)in_sizes; (void)n_in; (void)out_size; (void)ws_size;
  const float* x     = (const float*)d_in[0];   // (N, T, D)
  const float* A     = (const float*)d_in[1];   // (N, H, 4, 4) == (N, H, 16)
  const float* Wx    = (const float*)d_in[2];   // (D, 4H)
  const float* Wh    = (const float*)d_in[3];   // (H, 4H)
  const float* Wattn = (const float*)d_in[4];   // (H, 4H)
  const float* bb    = (const float*)d_in[5];   // (4H,)
  float* out = (float*)d_out;

  // workspace carve-out (all 256B-aligned)
  char* ws = (char*)d_ws;
  unsigned short* Wp = (unsigned short*)(ws);                    // 31,457,280 B
  unsigned short* Rp = (unsigned short*)(ws + 31457280);         //    786,432 B
  float* aout        = (float*)(ws + 32243712);                  // 10,485,760 B (4 split-K slices)
  float* hst         = (float*)(ws + 42729472);                  //    655,360 B
  float* cst         = (float*)(ws + 43384832);                  //    655,360 B

  pack_weights<<<(KTOT * NOUT + 255) / 256, 256, 0, stream>>>(Wx, Wh, Wattn, Wp);
  init_state<<<(NB * HH + 255) / 256, 256, 0, stream>>>(A, hst, cst, Rp);

  for (int t = 0; t < TT; ++t) {
    attn_step<<<NB, 256, 0, stream>>>(x, A, hst, Rp, t);
    gemm_step<<<dim3(NOUT / 64, KSPLIT), 256, 0, stream>>>(Rp, Wp, aout);
    gates_step<<<(NB * HH + 255) / 256, 256, 0, stream>>>(aout, bb, hst, cst, Rp, out, t);
  }
}
